// Model_49718541419161
// MI455X (gfx1250) — compile-verified, matrix operand order checked by reading, not run on hardware
//
// MI455X (gfx1250) implementation of the TimeMixer-style forecaster.
//
// Roofline: ~350 GFLOP of GEMM per forward on 32MB-class activations ->
// compute-bound on CDNA5; use V_WMMA_F32_16X16X32_BF16 (bf16 in, f32 acc).
// LDS staging via the Tensor Data Mover (tensor_load_to_lds + s_wait_tensorcnt),
// double-buffered so DMA overlaps WMMA. wave32, 8 waves/block, 128x64 tiles,
// 32x32 per wave. FFTs are lowered to DFT-matrix GEMMs so the whole model
// rides the WMMA path. residual = xa - seasonal - trend == 0 exactly ->
// residual embedding = tmark only.

#include <hip/hip_runtime.h>
#include <stdint.h>
#include <math.h>

typedef long long ll;
typedef __bf16 bf16;
typedef __attribute__((ext_vector_type(16))) __bf16   v16bf;
typedef __attribute__((ext_vector_type(8)))  float    v8f;
typedef __attribute__((ext_vector_type(4)))  uint32_t u32x4;
typedef __attribute__((ext_vector_type(8)))  int      i32x8;
typedef __attribute__((ext_vector_type(4)))  int      i32x4;

#define Bz 32
#define Tz 512
#define Nz 32
#define Dz 512
#define Hz 8
#define HDz 64
#define Fz 257
#define FKz 288   // 257 padded to 32 for WMMA K
#define PREDz 96
#define COz 32
#define M1z (Bz*Tz)

static inline int ceilDiv(int a, int b) { return (a + b - 1) / b; }

__device__ __forceinline__ float geluf(float x) {
  return 0.5f * x * (1.0f + erff(x * 0.70710678118f));
}

// ---------------------------------------------------------------------------
// Tensor Data Mover: 2D tile (32 elems along K, `rows` rows) global -> LDS.
// D# built per CDNA5 ISA (group0: count=1, lds_addr,
// global_addr, type=2; group1: data_size=2B, dims, tile dims, dim0 stride).
// ---------------------------------------------------------------------------
#if __has_builtin(__builtin_amdgcn_tensor_load_to_lds)
#define HAVE_TDM 1
__device__ __forceinline__ void tdm_load_tile(const bf16* g, uint32_t ldsOff,
                                              uint32_t rows, uint64_t strideElems) {
  uint64_t ga = (uint64_t)(uintptr_t)g;
  u32x4 g0;
  g0[0] = 1u;                                   // count = 1 valid descriptor
  g0[1] = ldsOff;                               // lds_addr
  g0[2] = (uint32_t)ga;                         // global_addr[31:0]
  g0[3] = (uint32_t)((ga >> 32) & 0x01FFFFFFu) | 0x80000000u; // addr[56:32] | type=2
  i32x8 g1;
  uint32_t d0 = 32u, d1 = rows;
  g1[0] = (int)(1u << 16);                      // data_size = 1 (2 bytes)
  g1[1] = (int)((d0 & 0xFFFFu) << 16);          // tensor_dim0 low16 -> bits63:48
  g1[2] = (int)(((d0 >> 16) & 0xFFFFu) | ((d1 & 0xFFFFu) << 16)); // dim0 hi / dim1 lo
  g1[3] = (int)(((d1 >> 16) & 0xFFFFu) | (32u << 16));            // dim1 hi / tile_dim0
  g1[4] = (int)(rows & 0xFFFFu);                // tile_dim1 (tile_dim2 = 0)
  g1[5] = (int)(uint32_t)(strideElems & 0xFFFFFFFFull);           // dim0_stride lo32
  g1[6] = (int)(uint32_t)((strideElems >> 32) & 0xFFFFull);       // dim0_stride hi16
  g1[7] = 0;
  i32x4 z4 = {0, 0, 0, 0};
#if __clang_major__ >= 23
  i32x8 z8 = {0, 0, 0, 0, 0, 0, 0, 0};
  __builtin_amdgcn_tensor_load_to_lds(g0, g1, z4, z4, z8, 0);
#else
  __builtin_amdgcn_tensor_load_to_lds(g0, g1, z4, z4, 0);
#endif
}
#endif

// ---------------------------------------------------------------------------
// Generic batched strided GEMM: C[b][m][n] = alpha * sum_k A[b][m][k]*B[b][n][k]
// (+bias)(+gelu)(+=C). A,B bf16 with K contiguous; C fp32. Batch index z is
// decomposed as outer*inner (e.g. batch x heads). 128x64 block tile, 8 waves,
// 32x32 per wave, K stepped by 32 via v_wmma_f32_16x16x32_bf16.
// TDM double-buffering: stage k+1 streams into the alternate LDS buffer while
// stage k is consumed; in-order TENSORcnt completion => wait tensorcnt<=2.
// The LDS object's address is escaped into an asm with a memory clobber so the
// compiler must assume the TDM wrote it (otherwise the fragment ds_loads are
// folded to undef, since no IR-visible store to the LDS arrays exists).
// Operand buffers are allocated with guard slack so over-read tile rows are
// harmless; out-of-range output rows/cols are masked in the epilogue.
// ---------------------------------------------------------------------------
struct GemmP {
  const bf16* A; const bf16* Bm; float* C;
  ll aRow, aInner, aOuter;
  ll bRow, bInner, bOuter;
  ll cRow, cInner, cOuter;
  const float* bias;
  int biasMode;          // 0 none, 1 per-n, 2 per-m
  int M, N, Kt, inner;   // Kt = K padded to 32
  int act;               // 0 none, 1 gelu
  int accum;             // 1: C += result
  float alpha;
};

#define LDS_A0 0u
#define LDS_A1 (128u * 32u * 2u)                 // 8192
#define LDS_B0 (2u * 128u * 32u * 2u)            // 16384
#define LDS_B1 (LDS_B0 + 64u * 32u * 2u)         // 20480

__global__ __launch_bounds__(256) void k_gemm(GemmP p) {
  __shared__ __align__(64) struct { bf16 A[2][128 * 32]; bf16 B[2][64 * 32]; } sm;
  const int tid  = threadIdx.x;
  const int wave = tid >> 5, lane = tid & 31;
  const int wm = wave & 3, wn = wave >> 2;
  const int m0 = blockIdx.x * 128, n0 = blockIdx.y * 64;
  const int bo = blockIdx.z / p.inner, bi = blockIdx.z % p.inner;
  const bf16* Ab = p.A  + (ll)bo * p.aOuter + (ll)bi * p.aInner + (ll)m0 * p.aRow;
  const bf16* Bb = p.Bm + (ll)bo * p.bOuter + (ll)bi * p.bInner + (ll)n0 * p.bRow;
  float* Cb = p.C + (ll)bo * p.cOuter + (ll)bi * p.cInner;
  const int nk = p.Kt >> 5;

  v8f acc[2][2] = {};
#ifdef HAVE_TDM
  if (wave == 0) {                                // prologue: stage 0 -> buffer 0
    tdm_load_tile(Ab, LDS_A0, 128u, (uint64_t)p.aRow);
    tdm_load_tile(Bb, LDS_B0, 64u, (uint64_t)p.bRow);
  }
  for (int ks = 0; ks < nk; ks++) {
    const int cur = ks & 1;
    if (wave == 0) {
      if (ks + 1 < nk) {                          // prefetch next stage
        const int nxt = (ks + 1) & 1;
        tdm_load_tile(Ab + (ll)(ks + 1) * 32, nxt ? LDS_A1 : LDS_A0, 128u, (uint64_t)p.aRow);
        tdm_load_tile(Bb + (ll)(ks + 1) * 32, nxt ? LDS_B1 : LDS_B0, 64u, (uint64_t)p.bRow);
#if __has_builtin(__builtin_amdgcn_s_wait_tensorcnt)
        __builtin_amdgcn_s_wait_tensorcnt(2);     // stage ks complete (in-order)
#endif
      } else {
#if __has_builtin(__builtin_amdgcn_s_wait_tensorcnt)
        __builtin_amdgcn_s_wait_tensorcnt(0);
#endif
      }
    }
    __syncthreads();
    // Escape the LDS object's address so the compiler must assume the TDM
    // (modeled by this asm's memory clobber) wrote it -> real ds_loads below.
    __asm__ volatile("" : : "v"((const void*)&sm) : "memory");
    const int fr = lane & 15, kb = (lane >> 4) * 16;
    v16bf a0 = *(const v16bf*)&sm.A[cur][(32 * wm +      fr) * 32 + kb];
    v16bf a1 = *(const v16bf*)&sm.A[cur][(32 * wm + 16 + fr) * 32 + kb];
    v16bf b0 = *(const v16bf*)&sm.B[cur][(32 * wn +      fr) * 32 + kb];
    v16bf b1 = *(const v16bf*)&sm.B[cur][(32 * wn + 16 + fr) * 32 + kb];
    acc[0][0] = __builtin_amdgcn_wmma_f32_16x16x32_bf16(false, a0, false, b0, (short)0, acc[0][0], false, false);
    acc[0][1] = __builtin_amdgcn_wmma_f32_16x16x32_bf16(false, a0, false, b1, (short)0, acc[0][1], false, false);
    acc[1][0] = __builtin_amdgcn_wmma_f32_16x16x32_bf16(false, a1, false, b0, (short)0, acc[1][0], false, false);
    acc[1][1] = __builtin_amdgcn_wmma_f32_16x16x32_bf16(false, a1, false, b1, (short)0, acc[1][1], false, false);
    __syncthreads();                              // done reading buf cur
  }
#else
  for (int ks = 0; ks < nk; ks++) {
    const int k0 = ks * 32;
    for (int i = tid; i < 128 * 2; i += 256) {    // A: 128 rows x 2 half-rows
      int r = i >> 1, h = (i & 1) * 16;
      for (int c = 0; c < 16; c++) sm.A[0][r * 32 + h + c] = Ab[(ll)r * p.aRow + k0 + h + c];
    }
    for (int i = tid; i < 64 * 2; i += 256) {     // B: 64 rows x 2 half-rows
      int r = i >> 1, h = (i & 1) * 16;
      for (int c = 0; c < 16; c++) sm.B[0][r * 32 + h + c] = Bb[(ll)r * p.bRow + k0 + h + c];
    }
    __syncthreads();
    const int fr = lane & 15, kb = (lane >> 4) * 16;
    v16bf a0 = *(const v16bf*)&sm.A[0][(32 * wm +      fr) * 32 + kb];
    v16bf a1 = *(const v16bf*)&sm.A[0][(32 * wm + 16 + fr) * 32 + kb];
    v16bf b0 = *(const v16bf*)&sm.B[0][(32 * wn +      fr) * 32 + kb];
    v16bf b1 = *(const v16bf*)&sm.B[0][(32 * wn + 16 + fr) * 32 + kb];
    acc[0][0] = __builtin_amdgcn_wmma_f32_16x16x32_bf16(false, a0, false, b0, (short)0, acc[0][0], false, false);
    acc[0][1] = __builtin_amdgcn_wmma_f32_16x16x32_bf16(false, a0, false, b1, (short)0, acc[0][1], false, false);
    acc[1][0] = __builtin_amdgcn_wmma_f32_16x16x32_bf16(false, a1, false, b0, (short)0, acc[1][0], false, false);
    acc[1][1] = __builtin_amdgcn_wmma_f32_16x16x32_bf16(false, a1, false, b1, (short)0, acc[1][1], false, false);
    __syncthreads();
  }
#endif
  for (int i = 0; i < 2; i++)
    for (int j = 0; j < 2; j++)
      for (int r = 0; r < 8; r++) {
        int gm = m0 + 32 * wm + 16 * i + r + 8 * (lane >> 4);
        int gn = n0 + 32 * wn + 16 * j + (lane & 15);
        if (gm < p.M && gn < p.N) {
          float v = p.alpha * acc[i][j][r];
          if (p.biasMode == 1) v += p.bias[gn];
          else if (p.biasMode == 2) v += p.bias[gm];
          if (p.act == 1) v = geluf(v);
          float* cp = Cb + (ll)gm * p.cRow + gn;
          if (p.accum) v += *cp;
          *cp = v;
        }
      }
}

// ---------------------------------------------------------------------------
// Support kernels (fp32 numerics; cheap relative to the GEMMs)
// ---------------------------------------------------------------------------
__global__ void k_stats(const float* x, const float* nw, const float* nb,
                        float* mean, float* rstd, float* mx) {
  int b = blockIdx.x / Nz, n = blockIdx.x % Nz;
  __shared__ float s1[256], s2[256];
  int t = threadIdx.x;
  float s = 0.f, sq = 0.f;
  for (int i = t; i < Tz; i += 256) { float v = x[((ll)b * Tz + i) * Nz + n]; s += v; sq += v * v; }
  s1[t] = s; s2[t] = sq; __syncthreads();
  for (int o = 128; o; o >>= 1) { if (t < o) { s1[t] += s1[t + o]; s2[t] += s2[t + o]; } __syncthreads(); }
  float m = s1[0] / Tz, var = s2[0] / Tz - (s1[0] / Tz) * (s1[0] / Tz);
  float rs = rsqrtf(var + 1e-5f);
  __syncthreads();
  float mv = -3.0e38f;
  for (int i = t; i < Tz; i += 256) {
    float v = (x[((ll)b * Tz + i) * Nz + n] - m) * rs * nw[n] + nb[n];
    mv = fmaxf(mv, v);
  }
  s1[t] = mv; __syncthreads();
  for (int o = 128; o; o >>= 1) { if (t < o) s1[t] = fmaxf(s1[t], s1[t + o]); __syncthreads(); }
  if (t == 0) { mean[blockIdx.x] = m; rstd[blockIdx.x] = rs; mx[blockIdx.x] = s1[0]; }
}

// SE channel attention. avg_t(xn) == norm_b exactly (mean of normalized is b).
__global__ void k_ca(const float* nb, const float* mx, const float* w1, const float* w2,
                     float* attw) {
  int b = blockIdx.x, n = threadIdx.x;  // 32 threads
  __shared__ float h[8];
  if (n < 8) {
    int j = n & 3; const float* v = (n < 4) ? nb : (mx + b * Nz);
    float a = 0.f;
    for (int m = 0; m < Nz; m++) a += w1[j * Nz + m] * v[m];
    h[n] = fmaxf(a, 0.f);
  }
  __syncthreads();
  float o = 0.f;
  for (int j = 0; j < 4; j++) o += w2[n * 4 + j] * (h[j] + h[j + 4]);
  attw[b * Nz + n] = 1.f / (1.f + expf(-o));
}

__global__ void k_decomp(const float* x, const float* mean, const float* rstd,
                         const float* nw, const float* nb, const float* attw,
                         float* seas, float* trnd) {
  int idx = blockIdx.x * 256 + threadIdx.x;
  if (idx >= Bz * Tz * Nz) return;
  int n = idx % Nz, t = (idx / Nz) % Tz, b = idx / (Nz * Tz);
  float m = mean[b * Nz + n], rs = rstd[b * Nz + n], aw = attw[b * Nz + n];
  float w = nw[n], bb = nb[n];
  float acc = 0.f;
  for (int j = -12; j <= 12; j++) {
    int tt = t + j; tt = tt < 0 ? 0 : (tt > Tz - 1 ? Tz - 1 : tt);
    acc += ((x[((ll)b * Tz + tt) * Nz + n] - m) * rs * w + bb) * aw;
  }
  float tr = acc * (1.f / 25.f);
  float xa = ((x[((ll)b * Tz + t) * Nz + n] - m) * rs * w + bb) * aw;
  trnd[idx] = tr;
  seas[idx] = xa - tr;
}

// token conv (k=3, circular) + time-mark embedding; seas==null -> tmark only.
__global__ void k_embed(const float* seas, const float* mark, const float* cw,
                        const float* tw, float* outF, bf16* outB) {
  int idx = blockIdx.x * 256 + threadIdx.x;
  if (idx >= M1z * Dz) return;
  int d = idx % Dz, row = idx / Dz;
  int t = row % Tz, b = row / Tz;
  float acc = 0.f;
  if (seas) {
    int tm = (t + Tz - 1) % Tz, tp = (t + 1) % Tz;
    const float* s0 = seas + ((ll)b * Tz + tm) * Nz;
    const float* s1 = seas + ((ll)b * Tz + t ) * Nz;
    const float* s2 = seas + ((ll)b * Tz + tp) * Nz;
    for (int n = 0; n < Nz; n++) {
      const float* w = cw + (d * Nz + n) * 3;
      acc += s0[n] * w[0] + s1[n] * w[1] + s2[n] * w[2];
    }
  }
  const float* mk = mark + ((ll)b * Tz + t) * 4;
  for (int j = 0; j < 4; j++) acc += mk[j] * tw[d * 4 + j];
  outF[idx] = acc;
  outB[idx] = (bf16)acc;
}

// strided fp32 -> bf16 with zero K-padding (also used for concat writes)
__global__ void k_cvt(const float* in, ll inRow, int C, bf16* out, ll outRow,
                      int off, int Cpad, int total) {
  int idx = blockIdx.x * 256 + threadIdx.x;
  if (idx >= total) return;
  int c = idx % Cpad, m = idx / Cpad;
  float v = (c < C) ? in[(ll)m * inRow + c] : 0.f;
  out[(ll)m * outRow + off + c] = (bf16)v;
}

// batched fp32 [b][R][C] (strided) -> bf16 [b][C][Rpad], zero pad beyond R
__global__ void k_transpose(const float* in, ll inRow, ll inInner, ll inOuter,
                            int inner, int R, int C, int Rpad, int total, bf16* out) {
  int idx = blockIdx.x * 256 + threadIdx.x;
  if (idx >= total) return;
  int r = idx % Rpad, rem = idx / Rpad;
  int c = rem % C, bidx = rem / C;
  int bo = bidx / inner, bi = bidx % inner;
  float v = (r < R) ? in[(ll)bo * inOuter + (ll)bi * inInner + (ll)r * inRow + c] : 0.f;
  out[((ll)bidx * C + c) * Rpad + r] = (bf16)v;
}

// Chebyshev basis: out[m][i*deg1+d] = cos(d*acos(tanh(x[m][i])))
__global__ void k_cheby(const float* x, int deg1, bf16* out) {
  int idx = blockIdx.x * 256 + threadIdx.x;
  if (idx >= M1z * Dz) return;
  int i = idx % Dz, m = idx / Dz;
  float th = acosf(tanhf(x[idx]));
  ll base = ((ll)m * Dz + i) * deg1;
  for (int d = 0; d < deg1; d++) out[base + d] = (bf16)cosf((float)d * th);
}

// coeffs [i][o][d] -> W[o][i*deg1+d]
__global__ void k_chebyw(const float* c, int deg1, bf16* out) {
  int idx = blockIdx.x * 256 + threadIdx.x;
  if (idx >= Dz * Dz * deg1) return;
  int d = idx % deg1, o = (idx / deg1) % Dz, i = idx / (deg1 * Dz);
  out[(ll)o * Dz * deg1 + (ll)i * deg1 + d] = (bf16)c[idx];
}

// mixer epilogue: LN(x + xk + dwconv(x)) over D=512 (one row per block)
__global__ void k_mixer_post(const float* x, const float* xk, const float* cw,
                             const float* cb, const float* lw, const float* lb,
                             float* out) {
  int row = blockIdx.x, t = row % Tz, tid = threadIdx.x;
  __shared__ float s1[256], s2[256];
  float v[2];
  for (int e = 0; e < 2; e++) {
    int d = tid + e * 256;
    float xc = x[(ll)row * Dz + d] * cw[d * 3 + 1] + cb[d];
    if (t > 0)      xc += x[(ll)(row - 1) * Dz + d] * cw[d * 3 + 0];
    if (t < Tz - 1) xc += x[(ll)(row + 1) * Dz + d] * cw[d * 3 + 2];
    v[e] = x[(ll)row * Dz + d] + xk[(ll)row * Dz + d] + xc;
  }
  s1[tid] = v[0] + v[1]; s2[tid] = v[0] * v[0] + v[1] * v[1];
  __syncthreads();
  for (int o = 128; o; o >>= 1) { if (tid < o) { s1[tid] += s1[tid + o]; s2[tid] += s2[tid + o]; } __syncthreads(); }
  float m = s1[0] / Dz, var = s2[0] / Dz - m * m, rs = rsqrtf(var + 1e-5f);
  for (int e = 0; e < 2; e++) {
    int d = tid + e * 256;
    out[(ll)row * Dz + d] = (v[e] - m) * rs * lw[d] + lb[d];
  }
}

// row softmax -> bf16 probs with zero K-pad (L <= 512)
__global__ void k_softmax(const float* in, int L, int Lk, bf16* out) {
  ll row = blockIdx.x;
  const float* r = in + row * (ll)L;
  __shared__ float sh[256];
  int t = threadIdx.x;
  float a = (t < L) ? r[t] : -3.0e38f;
  float b = (t + 256 < L) ? r[t + 256] : -3.0e38f;
  sh[t] = fmaxf(a, b); __syncthreads();
  for (int o = 128; o; o >>= 1) { if (t < o) sh[t] = fmaxf(sh[t], sh[t + o]); __syncthreads(); }
  float M = sh[0]; __syncthreads();
  float e0 = (t < L) ? expf(a - M) : 0.f;
  float e1 = (t + 256 < L) ? expf(b - M) : 0.f;
  sh[t] = e0 + e1; __syncthreads();
  for (int o = 128; o; o >>= 1) { if (t < o) sh[t] += sh[t + o]; __syncthreads(); }
  float inv = 1.f / sh[0];
  if (t < Lk)       out[row * Lk + t]       = (t < L)       ? (bf16)(e0 * inv) : (bf16)0.f;
  if (t + 256 < Lk) out[row * Lk + t + 256] = (t + 256 < L) ? (bf16)(e1 * inv) : (bf16)0.f;
}

__global__ void k_dft_fwd(bf16* Wr, bf16* Wi) {  // rFFT as GEMM: X_f = sum_t x_t e^{-2pi i f t/T}
  int idx = blockIdx.x * 256 + threadIdx.x;
  if (idx >= Fz * Tz) return;
  int t = idx % Tz, f = idx / Tz;
  float ang = 6.283185307179586f * (float)f * (float)t / (float)Tz;
  Wr[(ll)f * Tz + t] = (bf16)cosf(ang);
  Wi[(ll)f * Tz + t] = (bf16)(-sinf(ang));
}

__global__ void k_dft_inv(bf16* Wir, bf16* Wii) { // irfft weights incl. 1/T and hermitian 2x
  int idx = blockIdx.x * 256 + threadIdx.x;
  if (idx >= Tz * FKz) return;
  int f = idx % FKz, t = idx / FKz;
  float vr = 0.f, vi = 0.f;
  if (f < Fz) {
    float cf = (f == 0 || f == Fz - 1) ? 1.f : 2.f;
    float ang = 6.283185307179586f * (float)f * (float)t / (float)Tz;
    vr = cf * cosf(ang) / (float)Tz;
    vi = -cf * sinf(ang) / (float)Tz;
  }
  Wir[idx] = (bf16)vr;
  Wii[idx] = (bf16)vi;
}

__global__ void k_final_mix(const float* tf, const float* ff, const float* fu,
                            const float* bw, float* out) {
  int idx = blockIdx.x * 256 + threadIdx.x;
  if (idx >= M1z * Dz) return;
  float w0 = bw[0], w1 = bw[1], w2 = bw[2];
  float m = fmaxf(w0, fmaxf(w1, w2));
  float e0 = expf(w0 - m), e1 = expf(w1 - m), e2 = expf(w2 - m);
  float inv = 1.f / (e0 + e1 + e2);
  out[idx] = (e0 * tf[idx] + e1 * ff[idx] + e2 * fu[idx]) * inv;
}

__global__ void k_denorm(const float* pred, const float* nw, const float* nb,
                         const float* mean, const float* rstd, float* out) {
  int idx = blockIdx.x * 256 + threadIdx.x;
  if (idx >= Bz * PREDz * COz) return;
  int c = idx % COz, b = idx / (PREDz * COz);
  float v = pred[idx];
  v = (v - nb[c]) / (nw[c] + 1e-10f);
  out[idx] = v / rstd[b * Nz + c] + mean[b * Nz + c];
}

// ---------------------------------------------------------------------------
// Host orchestration
// ---------------------------------------------------------------------------
struct Arena {
  char* b; size_t o;
  void* take(size_t n) {            // 1MB guard: WMMA tiles deliberately over-read
    void* p = b + o;
    o += (n + (size_t)(1u << 20) + 255u) & ~(size_t)255u;
    return p;
  }
};

static void gemm_launch(hipStream_t s, const bf16* A, ll aR, ll aI, ll aO,
                        const bf16* B, ll bR, ll bI, ll bO,
                        float* C, ll cR, ll cI, ll cO,
                        int M, int N, int Kt, int batches, int inner,
                        const float* bias, int biasMode, int act, float alpha, int accum) {
  GemmP p;
  p.A = A; p.Bm = B; p.C = C;
  p.aRow = aR; p.aInner = aI; p.aOuter = aO;
  p.bRow = bR; p.bInner = bI; p.bOuter = bO;
  p.cRow = cR; p.cInner = cI; p.cOuter = cO;
  p.bias = bias; p.biasMode = biasMode;
  p.M = M; p.N = N; p.Kt = Kt; p.inner = inner;
  p.act = act; p.accum = accum; p.alpha = alpha;
  dim3 g(ceilDiv(M, 128), ceilDiv(N, 64), batches);
  k_gemm<<<g, 256, 0, s>>>(p);
}

static bf16* cvt_new(hipStream_t s, Arena& ar, const float* in, int M, int C, int Cpad) {
  bf16* out = (bf16*)ar.take((size_t)M * Cpad * 2);
  int n = M * Cpad;
  k_cvt<<<ceilDiv(n, 256), 256, 0, s>>>(in, (ll)C, C, out, (ll)Cpad, 0, Cpad, n);
  return out;
}
static void cvt_into(hipStream_t s, const float* in, int M, int C, bf16* out, ll outRow, int off) {
  int n = M * C;
  k_cvt<<<ceilDiv(n, 256), 256, 0, s>>>(in, (ll)C, C, out, outRow, off, C, n);
}

// MultiheadAttention (batch_first, eval): QKV gemm, scaled scores, softmax, PV, out proj
static void run_mha(hipStream_t s, Arena& ar, const bf16* xbf, int Bn, int L, int Lk,
                    const bf16* inw, const float* inb, const bf16* outw, const float* outb,
                    float* out) {
  int M = Bn * L;
  float* qkv = (float*)ar.take((size_t)M * 3 * Dz * 4);
  gemm_launch(s, xbf, Dz, 0, 0, inw, Dz, 0, 0, qkv, 3 * Dz, 0, 0,
              M, 3 * Dz, Dz, 1, 1, inb, 1, 0, 1.f, 0);
  bf16* qkvb = cvt_new(s, ar, qkv, M, 3 * Dz, 3 * Dz);
  float* sc = (float*)ar.take((size_t)Bn * Hz * L * L * 4);
  gemm_launch(s, qkvb, 3 * Dz, HDz, (ll)L * 3 * Dz,
                 qkvb + Dz, 3 * Dz, HDz, (ll)L * 3 * Dz,
                 sc, L, (ll)L * L, (ll)Hz * L * L,
              L, L, Dz, Bn * Hz, Hz, nullptr, 0, 0, 0.125f, 0);
  bf16* Pm = (bf16*)ar.take((size_t)Bn * Hz * L * Lk * 2);
  k_softmax<<<Bn * Hz * L, 256, 0, s>>>(sc, L, Lk, Pm);
  bf16* Vt = (bf16*)ar.take((size_t)Bn * Hz * HDz * Lk * 2);
  int tn = Bn * Hz * HDz * Lk;
  k_transpose<<<ceilDiv(tn, 256), 256, 0, s>>>(qkv + 2 * Dz, 3 * Dz, HDz, (ll)L * 3 * Dz,
                                               Hz, L, HDz, Lk, tn, Vt);
  float* O = (float*)ar.take((size_t)M * Dz * 4);
  gemm_launch(s, Pm, Lk, (ll)L * Lk, (ll)Hz * L * Lk,
                 Vt, Lk, (ll)HDz * Lk, (ll)Hz * HDz * Lk,
                 O, Dz, HDz, (ll)L * Dz,
              L, HDz, Lk, Bn * Hz, Hz, nullptr, 0, 0, 1.f, 0);
  bf16* Ob = cvt_new(s, ar, O, M, Dz, Dz);
  gemm_launch(s, Ob, Dz, 0, 0, outw, Dz, 0, 0, out, Dz, 0, 0,
              M, Dz, Dz, 1, 1, outb, 1, 0, 1.f, 0);
}

static void run_mixer(hipStream_t s, Arena& ar, void* const* d_in, int pb, int deg1,
                      const float* embF, bf16* catT, int slot) {
  auto P = [&](int i) { return (const float*)d_in[i]; };
  size_t mk = ar.o;
  int K = Dz * deg1;
  bf16* basis = (bf16*)ar.take((size_t)M1z * K * 2);
  k_cheby<<<ceilDiv(M1z * Dz, 256), 256, 0, s>>>(embF, deg1, basis);
  bf16* wk = (bf16*)ar.take((size_t)Dz * K * 2);
  k_chebyw<<<ceilDiv(Dz * K, 256), 256, 0, s>>>(P(pb + 0), deg1, wk);
  float* xk = (float*)ar.take((size_t)M1z * Dz * 4);
  gemm_launch(s, basis, K, 0, 0, wk, K, 0, 0, xk, Dz, 0, 0,
              M1z, Dz, K, 1, 1, nullptr, 0, 0, 1.f, 0);
  float* mo = (float*)ar.take((size_t)M1z * Dz * 4);
  k_mixer_post<<<M1z, 256, 0, s>>>(embF, xk, P(pb + 1), P(pb + 2), P(pb + 3), P(pb + 4), mo);
  cvt_into(s, mo, M1z, Dz, catT, 3 * Dz, slot * Dz);
  ar.o = mk;
}

static void run_fourier(hipStream_t s, Arena& ar, void* const* d_in, int pb,
                        const bf16* Wr, const bf16* Wi, const bf16* Wir, const bf16* Wii,
                        const bf16* embB, bf16* catF, int slot) {
  auto P = [&](int i) { return (const float*)d_in[i]; };
  size_t mk = ar.o;
  bf16* mapw = cvt_new(s, ar, P(pb + 0), Dz, Dz, Dz);
  bf16* adw  = cvt_new(s, ar, P(pb + 2), Dz, 2 * Dz, 2 * Dz);
  bf16* inw  = cvt_new(s, ar, P(pb + 4), 3 * Dz, Dz, Dz);
  bf16* outw = cvt_new(s, ar, P(pb + 6), Dz, Dz, Dz);
  bf16* f1w  = cvt_new(s, ar, P(pb + 8), 2 * Dz, Dz, Dz);
  bf16* f2w  = cvt_new(s, ar, P(pb + 10), Dz, 2 * Dz, 2 * Dz);
  bf16* oaw  = cvt_new(s, ar, P(pb + 12), Dz, Dz / 2, Dz / 2);
  bf16* prw  = cvt_new(s, ar, P(pb + 14), Dz, Dz, Dz);

  float* xm = (float*)ar.take((size_t)M1z * Dz * 4);
  gemm_launch(s, embB, Dz, 0, 0, mapw, Dz, 0, 0, xm, Dz, 0, 0,
              M1z, Dz, Dz, 1, 1, P(pb + 1), 1, 0, 1.f, 0);
  bf16* xmT = (bf16*)ar.take((size_t)Bz * Dz * Tz * 2);
  int tn = Bz * Dz * Tz;
  k_transpose<<<ceilDiv(tn, 256), 256, 0, s>>>(xm, Dz, 0, (ll)Tz * Dz, 1, Tz, Dz, Tz, tn, xmT);

  int MF = Bz * Fz;
  float* xr = (float*)ar.take((size_t)MF * 2 * Dz * 4);
  gemm_launch(s, Wr, Tz, 0, 0, xmT, Tz, 0, (ll)Dz * Tz, xr, 2 * Dz, 0, (ll)Fz * 2 * Dz,
              Fz, Dz, Tz, Bz, 1, nullptr, 0, 0, 1.f, 0);
  gemm_launch(s, Wi, Tz, 0, 0, xmT, Tz, 0, (ll)Dz * Tz, xr + Dz, 2 * Dz, 0, (ll)Fz * 2 * Dz,
              Fz, Dz, Tz, Bz, 1, nullptr, 0, 0, 1.f, 0);
  bf16* xrb = cvt_new(s, ar, xr, MF, 2 * Dz, 2 * Dz);
  float* xa = (float*)ar.take((size_t)MF * Dz * 4);
  gemm_launch(s, xrb, 2 * Dz, 0, 0, adw, 2 * Dz, 0, 0, xa, Dz, 0, 0,
              MF, Dz, 2 * Dz, 1, 1, P(pb + 3), 1, 0, 1.f, 0);
  bf16* xab = cvt_new(s, ar, xa, MF, Dz, Dz);
  float* att = (float*)ar.take((size_t)MF * Dz * 4);
  run_mha(s, ar, xab, Bz, Fz, FKz, inw, P(pb + 5), outw, P(pb + 7), att);
  bf16* attb = cvt_new(s, ar, att, MF, Dz, Dz);
  float* f1 = (float*)ar.take((size_t)MF * 2 * Dz * 4);
  gemm_launch(s, attb, Dz, 0, 0, f1w, Dz, 0, 0, f1, 2 * Dz, 0, 0,
              MF, 2 * Dz, Dz, 1, 1, P(pb + 9), 1, 1 /*gelu*/, 1.f, 0);
  bf16* f1b = cvt_new(s, ar, f1, MF, 2 * Dz, 2 * Dz);
  float* f2 = (float*)ar.take((size_t)MF * Dz * 4);
  gemm_launch(s, f1b, 2 * Dz, 0, 0, f2w, 2 * Dz, 0, 0, f2, Dz, 0, 0,
              MF, Dz, 2 * Dz, 1, 1, P(pb + 11), 1, 0, 1.f, 0);

  bf16* frT = (bf16*)ar.take((size_t)Bz * 256 * FKz * 2);
  bf16* fiT = (bf16*)ar.take((size_t)Bz * 256 * FKz * 2);
  tn = Bz * 256 * FKz;
  k_transpose<<<ceilDiv(tn, 256), 256, 0, s>>>(f2,       Dz, 0, (ll)Fz * Dz, 1, Fz, 256, FKz, tn, frT);
  k_transpose<<<ceilDiv(tn, 256), 256, 0, s>>>(f2 + 256, Dz, 0, (ll)Fz * Dz, 1, Fz, 256, FKz, tn, fiT);
  float* xi = (float*)ar.take((size_t)M1z * 256 * 4);
  gemm_launch(s, Wir, FKz, 0, 0, frT, FKz, 0, (ll)256 * FKz, xi, 256, 0, (ll)Tz * 256,
              Tz, 256, FKz, Bz, 1, nullptr, 0, 0, 1.f, 0);
  gemm_launch(s, Wii, FKz, 0, 0, fiT, FKz, 0, (ll)256 * FKz, xi, 256, 0, (ll)Tz * 256,
              Tz, 256, FKz, Bz, 1, nullptr, 0, 0, 1.f, 1 /*accum*/);
  bf16* xib = cvt_new(s, ar, xi, M1z, 256, 256);
  float* xo = (float*)ar.take((size_t)M1z * Dz * 4);
  gemm_launch(s, xib, 256, 0, 0, oaw, 256, 0, 0, xo, Dz, 0, 0,
              M1z, Dz, 256, 1, 1, P(pb + 13), 1, 0, 1.f, 0);
  bf16* xob = cvt_new(s, ar, xo, M1z, Dz, Dz);
  float* fb = (float*)ar.take((size_t)M1z * Dz * 4);
  gemm_launch(s, xob, Dz, 0, 0, prw, Dz, 0, 0, fb, Dz, 0, 0,
              M1z, Dz, Dz, 1, 1, P(pb + 15), 1, 0, 1.f, 0);
  cvt_into(s, fb, M1z, Dz, catF, 3 * Dz, slot * Dz);
  ar.o = mk;
}

extern "C" void kernel_launch(void* const* d_in, const int* in_sizes, int n_in,
                              void* d_out, int out_size, void* d_ws, size_t ws_size,
                              hipStream_t stream) {
  (void)in_sizes; (void)n_in; (void)out_size; (void)ws_size;
  auto P = [&](int i) { return (const float*)d_in[i]; };
  Arena ar{(char*)d_ws, 0};
  const float* x_enc = P(0);
  const float* x_mark = P(1);
  const float* nw = P(2);
  const float* nb = P(3);

  // ---- RevIN stats, channel attention, series decomposition -------------
  float* mean = (float*)ar.take(Bz * Nz * 4);
  float* rstd = (float*)ar.take(Bz * Nz * 4);
  float* mx   = (float*)ar.take(Bz * Nz * 4);
  float* attw = (float*)ar.take(Bz * Nz * 4);
  k_stats<<<Bz * Nz, 256, 0, stream>>>(x_enc, nw, nb, mean, rstd, mx);
  k_ca<<<Bz, 32, 0, stream>>>(nb, mx, P(4), P(5), attw);
  float* seas = (float*)ar.take((size_t)Bz * Tz * Nz * 4);
  float* trnd = (float*)ar.take((size_t)Bz * Tz * Nz * 4);
  k_decomp<<<ceilDiv(Bz * Tz * Nz, 256), 256, 0, stream>>>(x_enc, mean, rstd, nw, nb, attw, seas, trnd);

  // ---- embeddings: trend, seasonal, residual(==0 -> tmark only) ---------
  const float* srcs[3] = {trnd, seas, nullptr};
  float* embF[3]; bf16* embB[3];
  for (int i = 0; i < 3; i++) {
    embF[i] = (float*)ar.take((size_t)M1z * Dz * 4);
    embB[i] = (bf16*)ar.take((size_t)M1z * Dz * 2);
    k_embed<<<ceilDiv(M1z * Dz, 256), 256, 0, stream>>>(srcs[i], x_mark, P(6), P(7), embF[i], embB[i]);
  }

  // ---- DFT matrices (FFT lowered to WMMA GEMMs) -------------------------
  bf16* Wr  = (bf16*)ar.take((size_t)384 * Tz * 2);
  bf16* Wi  = (bf16*)ar.take((size_t)384 * Tz * 2);
  bf16* Wir = (bf16*)ar.take((size_t)Tz * FKz * 2);
  bf16* Wii = (bf16*)ar.take((size_t)Tz * FKz * 2);
  k_dft_fwd<<<ceilDiv(Fz * Tz, 256), 256, 0, stream>>>(Wr, Wi);
  k_dft_inv<<<ceilDiv(Tz * FKz, 256), 256, 0, stream>>>(Wir, Wii);

  // persistent branch outputs
  bf16*  catT   = (bf16*)ar.take((size_t)M1z * 3 * Dz * 2);
  bf16*  catF   = (bf16*)ar.take((size_t)M1z * 3 * Dz * 2);
  float* timeF  = (float*)ar.take((size_t)M1z * Dz * 4);
  float* freqF  = (float*)ar.take((size_t)M1z * Dz * 4);
  bf16*  catTF  = (bf16*)ar.take((size_t)M1z * 2 * Dz * 2);
  float* fusedF = (float*)ar.take((size_t)M1z * Dz * 4);
  float* finalF = (float*)ar.take((size_t)M1z * Dz * 4);

  // ---- time branch: 3 ChebyKAN mixers -> concat -> linear ---------------
  run_mixer(stream, ar, d_in,  8, 4, embF[0], catT, 0);  // trend (deg 3)
  run_mixer(stream, ar, d_in, 13, 6, embF[1], catT, 1);  // seasonal (deg 5)
  run_mixer(stream, ar, d_in, 18, 5, embF[2], catT, 2);  // residual (deg 4)
  {
    size_t mk = ar.o;
    bf16* tfw = cvt_new(stream, ar, P(23), Dz, 3 * Dz, 3 * Dz);
    gemm_launch(stream, catT, 3 * Dz, 0, 0, tfw, 3 * Dz, 0, 0, timeF, Dz, 0, 0,
                M1z, Dz, 3 * Dz, 1, 1, P(24), 1, 0, 1.f, 0);
    ar.o = mk;
  }

  // ---- freq branch: 3 fourier blocks -> concat -> MLP -------------------
  run_fourier(stream, ar, d_in, 25, Wr, Wi, Wir, Wii, embB[0], catF, 0);
  run_fourier(stream, ar, d_in, 41, Wr, Wi, Wir, Wii, embB[1], catF, 1);
  run_fourier(stream, ar, d_in, 57, Wr, Wi, Wir, Wii, embB[2], catF, 2);
  {
    size_t mk = ar.o;
    bf16* f1w = cvt_new(stream, ar, P(73), 2 * Dz, 3 * Dz, 3 * Dz);
    bf16* f2w = cvt_new(stream, ar, P(75), Dz, 2 * Dz, 2 * Dz);
    float* g1 = (float*)ar.take((size_t)M1z * 2 * Dz * 4);
    gemm_launch(stream, catF, 3 * Dz, 0, 0, f1w, 3 * Dz, 0, 0, g1, 2 * Dz, 0, 0,
                M1z, 2 * Dz, 3 * Dz, 1, 1, P(74), 1, 1 /*gelu*/, 1.f, 0);
    bf16* g1b = cvt_new(stream, ar, g1, M1z, 2 * Dz, 2 * Dz);
    gemm_launch(stream, g1b, 2 * Dz, 0, 0, f2w, 2 * Dz, 0, 0, freqF, Dz, 0, 0,
                M1z, Dz, 2 * Dz, 1, 1, P(76), 1, 0, 1.f, 0);
    ar.o = mk;
  }

  // ---- fusion: 1x1 conv over [time|freq] + self attention over T --------
  {
    size_t mk = ar.o;
    cvt_into(stream, timeF, M1z, Dz, catTF, 2 * Dz, 0);
    cvt_into(stream, freqF, M1z, Dz, catTF, 2 * Dz, Dz);
    bf16* fcw = cvt_new(stream, ar, P(77), Dz, 2 * Dz, 2 * Dz);
    float* f0 = (float*)ar.take((size_t)M1z * Dz * 4);
    gemm_launch(stream, catTF, 2 * Dz, 0, 0, fcw, 2 * Dz, 0, 0, f0, Dz, 0, 0,
                M1z, Dz, 2 * Dz, 1, 1, P(78), 1, 0, 1.f, 0);
    bf16* f0b = cvt_new(stream, ar, f0, M1z, Dz, Dz);
    bf16* inw  = cvt_new(stream, ar, P(79), 3 * Dz, Dz, Dz);
    bf16* outw = cvt_new(stream, ar, P(81), Dz, Dz, Dz);
    run_mha(stream, ar, f0b, Bz, Tz, Tz, inw, P(80), outw, P(82), fusedF);
    ar.o = mk;
  }

  // ---- weighted mix, temporal predictor, projection, denorm -------------
  k_final_mix<<<ceilDiv(M1z * Dz, 256), 256, 0, stream>>>(timeF, freqF, fusedF, P(83), finalF);
  {
    size_t mk = ar.o;
    bf16* finalT = (bf16*)ar.take((size_t)Bz * Dz * Tz * 2);
    int tn = Bz * Dz * Tz;
    k_transpose<<<ceilDiv(tn, 256), 256, 0, stream>>>(finalF, Dz, 0, (ll)Tz * Dz, 1,
                                                      Tz, Dz, Tz, tn, finalT);
    bf16* tpw = cvt_new(stream, ar, P(84), PREDz, Tz, Tz);
    float* pred = (float*)ar.take((size_t)Bz * PREDz * Dz * 4);
    gemm_launch(stream, tpw, Tz, 0, 0, finalT, Tz, 0, (ll)Dz * Tz,
                pred, Dz, 0, (ll)PREDz * Dz,
                PREDz, Dz, Tz, Bz, 1, P(85), 2 /*per-m bias*/, 0, 1.f, 0);
    bf16* predb = cvt_new(stream, ar, pred, Bz * PREDz, Dz, Dz);
    bf16* pjw = cvt_new(stream, ar, P(86), COz, Dz, Dz);
    float* dec = (float*)ar.take((size_t)Bz * PREDz * COz * 4);
    gemm_launch(stream, predb, Dz, 0, 0, pjw, Dz, 0, 0, dec, COz, 0, 0,
                Bz * PREDz, COz, Dz, 1, 1, P(87), 1, 0, 1.f, 0);
    k_denorm<<<ceilDiv(Bz * PREDz * COz, 256), 256, 0, stream>>>(dec, nw, nb, mean, rstd,
                                                                 (float*)d_out);
    ar.o = mk;
  }
}
